// PhaseLinear_85882166051585
// MI455X (gfx1250) — matrix operand order repaired and checked by reading, not compiled, not run on hardware
//
#include <hip/hip_runtime.h>
#include <hip/hip_bf16.h>

typedef __attribute__((ext_vector_type(16))) __bf16 v16bf;
typedef __attribute__((ext_vector_type(8)))  float  v8f;
typedef int v4i_b128 __attribute__((vector_size(16)));

#define B_DIM   4096
#define IN_DIM  1024
#define OUT_DIM 1024
#define BK 32
#define KST 40   // LDS row stride in bf16 elements (80 B: 16B-aligned, conflict-free)

// Workspace layout (all bf16 planes):
//   [0,        8MB)  W_hi   (4*1024*1024)
//   [8MB,     16MB)  W_lo
//   [16MB,    24MB)  X_hi   (4096*1024)
//   [24MB,    32MB)  X_lo
#define W_ELEMS (4L * OUT_DIM * IN_DIM)
#define X_ELEMS ((long)B_DIM * IN_DIM)
#define WS_NEED (2L * (W_ELEMS + X_ELEMS) * 2)   // 33,554,432 bytes

#if __has_builtin(__builtin_amdgcn_global_load_async_to_lds_b128)
#define ASYNC_LDS 1
#else
#define ASYNC_LDS 0
#endif

#define AS1 __attribute__((address_space(1)))
#define AS3 __attribute__((address_space(3)))

// Stage 16 bytes global -> LDS. CDNA5 async DMA path (ASYNCcnt-tracked,
// no VGPR round-trip) when available; plain copy otherwise.
__device__ __forceinline__ void stage16(const unsigned short* __restrict__ g,
                                        unsigned short* l) {
#if ASYNC_LDS
    __builtin_amdgcn_global_load_async_to_lds_b128(
        (AS1 v4i_b128*)g, (AS3 v4i_b128*)l, 0, 0);
#else
    *(uint4*)l = *(const uint4*)g;
#endif
}

__device__ __forceinline__ void stage_fence() {
#if ASYNC_LDS
#if __has_builtin(__builtin_amdgcn_s_wait_asynccnt)
    __builtin_amdgcn_s_wait_asynccnt(0);
#else
    asm volatile("s_wait_asynccnt 0x0" ::: "memory");
#endif
#endif
}

// Split fp32 into bf16 hi + bf16 lo (truncation split; hi+lo carries ~16 mantissa bits)
__device__ __forceinline__ void split2(float x0, float x1, unsigned& hi, unsigned& lo) {
    unsigned u0 = __float_as_uint(x0);
    unsigned u1 = __float_as_uint(x1);
    unsigned h0 = u0 >> 16, h1 = u1 >> 16;
    float r0 = x0 - __uint_as_float(h0 << 16);
    float r1 = x1 - __uint_as_float(h1 << 16);
    unsigned l0 = __float_as_uint(r0) >> 16;
    unsigned l1 = __float_as_uint(r1) >> 16;
    hi = h0 | (h1 << 16);
    lo = l0 | (l1 << 16);
}

union Frag {
    v16bf v;
    uint4 q[2];
};

// ---------------- pre-conversion kernel (memory bound, one pass) ----------------
__global__ __launch_bounds__(256)
void split_convert_kernel(const float* __restrict__ src,
                          unsigned short* __restrict__ hi,
                          unsigned short* __restrict__ lo, int n4) {
    int i = blockIdx.x * blockDim.x + threadIdx.x;
    if (i >= n4) return;
    float4 v = ((const float4*)src)[i];
    uint2 h, l;
    split2(v.x, v.y, h.x, l.x);
    split2(v.z, v.w, h.y, l.y);
    ((uint2*)hi)[i] = h;
    ((uint2*)lo)[i] = l;
}

// ---------------- shared epilogue helper ----------------
__device__ __forceinline__ void spline_eff(float ph, float& e0, float& e1,
                                           float& e2, float& e3) {
    const float INV_HALF_PI = 0.63661977236758134308f; // 2/pi
    const float s = ph * INV_HALF_PI;
    int q = (int)floorf(s);
    q = q < 0 ? 0 : (q > 3 ? 3 : q);
    const float t  = s - (float)q;
    const float t2 = t * t;
    const float t3 = t2 * t;
    const float c0 = -0.5f * t3 +        t2 - 0.5f * t;
    const float c1 =  1.5f * t3 - 2.5f * t2 + 1.0f;
    const float c2 = -1.5f * t3 + 2.0f * t2 + 0.5f * t;
    const float c3 =  0.5f * t3 - 0.5f * t2;
    // eff[cp] = coeff[(cp + 1 - q) mod 4]  (CPI rows are rotations)
    e0 = (q == 0) ? c1 : (q == 1) ? c0 : (q == 2) ? c3 : c2;
    e1 = (q == 0) ? c2 : (q == 1) ? c1 : (q == 2) ? c0 : c3;
    e2 = (q == 0) ? c3 : (q == 1) ? c2 : (q == 2) ? c1 : c0;
    e3 = (q == 0) ? c0 : (q == 1) ? c3 : (q == 2) ? c2 : c1;
}

// ================= fast path: pre-converted bf16 planes, BM=128 =================
// block = 256 threads (8 wave32s, 4(M) x 2(N)); each wave computes 32x32.
// Double-buffered LDS (123 KB of the WGP's 320 KB): async DMA of tile i+1
// overlaps the 48 WMMAs of tile i.
__global__ __launch_bounds__(256)
void PhaseLinear_wmma_pre(const unsigned short* __restrict__ Xhi,
                          const unsigned short* __restrict__ Xlo,
                          const unsigned short* __restrict__ Whi_g,
                          const unsigned short* __restrict__ Wlo_g,
                          const float* __restrict__ phase,
                          const float* __restrict__ biases,
                          float* __restrict__ out)
{
    constexpr int BM = 128, BN = 64;
    __shared__ alignas(16) unsigned short Ahi[2][BM * KST];
    __shared__ alignas(16) unsigned short Alo[2][BM * KST];
    __shared__ alignas(16) unsigned short Whi[2][4][BN * KST];
    __shared__ alignas(16) unsigned short Wlo[2][4][BN * KST];

    const int n_blk = blockIdx.x * BN;
    const int m_blk = blockIdx.y * BM;

    const int tid  = threadIdx.x;
    const int lane = tid & 31;
    const int wv   = tid >> 5;
    const int wm   = wv & 3;   // 4 waves along M (32 rows each)
    const int wn   = wv >> 2;  // 2 waves along N (32 cols each)

    // staging roles
    const int arow_s = tid >> 1;        // 0..127 (A rows)
    const int aq     = (tid & 1) * 16;  // 0 or 16 halves
    const int wrow_s = tid >> 2;        // 0..63  (W rows)
    const int wq     = (tid & 3) * 8;   // 0,8,16,24 halves

    v8f acc[4][2][2];  // [cp][mt][nt]
    #pragma unroll
    for (int c = 0; c < 4; ++c)
        #pragma unroll
        for (int mt = 0; mt < 2; ++mt)
            #pragma unroll
            for (int nt = 0; nt < 2; ++nt)
                #pragma unroll
                for (int i = 0; i < 8; ++i)
                    acc[c][mt][nt][i] = 0.0f;

    const unsigned short* xh_row = Xhi + (long)(m_blk + arow_s) * IN_DIM + aq;
    const unsigned short* xl_row = Xlo + (long)(m_blk + arow_s) * IN_DIM + aq;

    auto stage_tile = [&](int buf, int kc) {
        stage16(xh_row + kc,     &Ahi[buf][arow_s * KST + aq]);
        stage16(xh_row + kc + 8, &Ahi[buf][arow_s * KST + aq + 8]);
        stage16(xl_row + kc,     &Alo[buf][arow_s * KST + aq]);
        stage16(xl_row + kc + 8, &Alo[buf][arow_s * KST + aq + 8]);
        #pragma unroll
        for (int c = 0; c < 4; ++c) {
            const long off = ((long)c * OUT_DIM + n_blk + wrow_s) * IN_DIM + kc + wq;
            stage16(Whi_g + off, &Whi[buf][c][wrow_s * KST + wq]);
            stage16(Wlo_g + off, &Wlo[buf][c][wrow_s * KST + wq]);
        }
    };

    constexpr int NITER = IN_DIM / BK;  // 32
    stage_tile(0, 0);                   // prologue DMA -> buf0

    for (int it = 0; it < NITER; ++it) {
        const int cur = it & 1;
        stage_fence();       // my DMAs for tile 'it' complete (only ones outstanding)
        __syncthreads();     // all waves' DMAs done; all reads of buf[cur^1] done
        if (it + 1 < NITER)
            stage_tile(cur ^ 1, (it + 1) * BK);  // overlaps with compute below

        // ---- A fragments for both 16-row sub-tiles ----
        const int kb = (lane < 16) ? 0 : 8;
        Frag ah[2], al[2];
        #pragma unroll
        for (int mt = 0; mt < 2; ++mt) {
            const int arow = wm * 32 + mt * 16 + (lane & 15);
            const unsigned short* ab  = &Ahi[cur][arow * KST];
            const unsigned short* alb = &Alo[cur][arow * KST];
            ah[mt].q[0] = *(const uint4*)(ab  + kb);
            ah[mt].q[1] = *(const uint4*)(ab  + kb + 16);
            al[mt].q[0] = *(const uint4*)(alb + kb);
            al[mt].q[1] = *(const uint4*)(alb + kb + 16);
        }

        const int koff = (lane < 16) ? 0 : 8;
        #pragma unroll
        for (int nt = 0; nt < 2; ++nt) {
            const int nloc = wn * 32 + nt * 16 + (lane & 15);
            #pragma unroll
            for (int c = 0; c < 4; ++c) {
                Frag bh, bl;
                const unsigned short* wb  = &Whi[cur][c][nloc * KST];
                const unsigned short* wlb = &Wlo[cur][c][nloc * KST];
                bh.q[0] = *(const uint4*)(wb  + koff);
                bh.q[1] = *(const uint4*)(wb  + koff + 16);
                bl.q[0] = *(const uint4*)(wlb + koff);
                bl.q[1] = *(const uint4*)(wlb + koff + 16);
                #pragma unroll
                for (int mt = 0; mt < 2; ++mt) {
                    acc[c][mt][nt] = __builtin_amdgcn_wmma_f32_16x16x32_bf16(
                        false, ah[mt].v, false, bh.v, (short)0, acc[c][mt][nt], false, false);
                    acc[c][mt][nt] = __builtin_amdgcn_wmma_f32_16x16x32_bf16(
                        false, ah[mt].v, false, bl.v, (short)0, acc[c][mt][nt], false, false);
                    acc[c][mt][nt] = __builtin_amdgcn_wmma_f32_16x16x32_bf16(
                        false, al[mt].v, false, bh.v, (short)0, acc[c][mt][nt], false, false);
                }
            }
        }
    }

    // ---- epilogue ----
    const int colLane = lane & 15;
    const int mTop    = (lane >> 4) << 3;

    float bb[2][4];
    #pragma unroll
    for (int nt = 0; nt < 2; ++nt) {
        const int ng = n_blk + wn * 32 + nt * 16 + colLane;
        #pragma unroll
        for (int c = 0; c < 4; ++c) bb[nt][c] = biases[c * OUT_DIM + ng];
    }

    #pragma unroll
    for (int mt = 0; mt < 2; ++mt) {
        #pragma unroll
        for (int r = 0; r < 8; ++r) {
            const int rowg = m_blk + wm * 32 + mt * 16 + mTop + r;
            float e0, e1, e2, e3;
            spline_eff(phase[rowg], e0, e1, e2, e3);
            #pragma unroll
            for (int nt = 0; nt < 2; ++nt) {
                const int ng = n_blk + wn * 32 + nt * 16 + colLane;
                const float v = e0 * (acc[0][mt][nt][r] + bb[nt][0])
                              + e1 * (acc[1][mt][nt][r] + bb[nt][1])
                              + e2 * (acc[2][mt][nt][r] + bb[nt][2])
                              + e3 * (acc[3][mt][nt][r] + bb[nt][3]);
                out[(long)rowg * OUT_DIM + ng] = v;
            }
        }
    }
}

// ================= fallback path: on-the-fly conversion, BM=64 =================
__global__ __launch_bounds__(256)
void PhaseLinear_wmma_fly(const float* __restrict__ x,
                          const float* __restrict__ phase,
                          const float* __restrict__ wts,
                          const float* __restrict__ biases,
                          float* __restrict__ out)
{
    constexpr int BM = 64, BN = 64;
    __shared__ alignas(16) unsigned short Ahi[BM * KST];
    __shared__ alignas(16) unsigned short Alo[BM * KST];
    __shared__ alignas(16) unsigned short Whi[4][BN * KST];
    __shared__ alignas(16) unsigned short Wlo[4][BN * KST];

    const int n_blk = blockIdx.x * BN;
    const int m_blk = blockIdx.y * BM;

    const int tid  = threadIdx.x;
    const int lane = tid & 31;
    const int wv   = tid >> 5;
    const int wm   = wv & 3;
    const int wn   = wv >> 2;

    const int srow = tid >> 2;
    const int sq   = (tid & 3) * 8;

    v8f acc[4][2];
    #pragma unroll
    for (int c = 0; c < 4; ++c)
        #pragma unroll
        for (int nt = 0; nt < 2; ++nt)
            #pragma unroll
            for (int i = 0; i < 8; ++i)
                acc[c][nt][i] = 0.0f;

    const float* xrow = x + (long)(m_blk + srow) * IN_DIM + sq;

    for (int kc = 0; kc < IN_DIM; kc += BK) {
        __syncthreads();
        {
            float4 a0 = *(const float4*)(xrow + kc);
            float4 a1 = *(const float4*)(xrow + kc + 4);
            uint4 h, l;
            split2(a0.x, a0.y, h.x, l.x);
            split2(a0.z, a0.w, h.y, l.y);
            split2(a1.x, a1.y, h.z, l.z);
            split2(a1.z, a1.w, h.w, l.w);
            *(uint4*)&Ahi[srow * KST + sq] = h;
            *(uint4*)&Alo[srow * KST + sq] = l;
        }
        #pragma unroll
        for (int c = 0; c < 4; ++c) {
            const float* wp = wts + ((long)c * OUT_DIM + n_blk + srow) * IN_DIM + kc + sq;
            float4 w0 = *(const float4*)(wp);
            float4 w1 = *(const float4*)(wp + 4);
            uint4 wh, wl;
            split2(w0.x, w0.y, wh.x, wl.x);
            split2(w0.z, w0.w, wh.y, wl.y);
            split2(w1.x, w1.y, wh.z, wl.z);
            split2(w1.z, w1.w, wh.w, wl.w);
            *(uint4*)&Whi[c][srow * KST + sq] = wh;
            *(uint4*)&Wlo[c][srow * KST + sq] = wl;
        }
        __syncthreads();

        const int arow = wm * 16 + (lane & 15);
        const int kb   = (lane < 16) ? 0 : 8;
        Frag ah, al;
        {
            const unsigned short* ab  = &Ahi[arow * KST];
            const unsigned short* alb = &Alo[arow * KST];
            ah.q[0] = *(const uint4*)(ab  + kb);
            ah.q[1] = *(const uint4*)(ab  + kb + 16);
            al.q[0] = *(const uint4*)(alb + kb);
            al.q[1] = *(const uint4*)(alb + kb + 16);
        }

        const int koff = (lane < 16) ? 0 : 8;
        #pragma unroll
        for (int nt = 0; nt < 2; ++nt) {
            const int nloc = wn * 32 + nt * 16 + (lane & 15);
            #pragma unroll
            for (int c = 0; c < 4; ++c) {
                Frag bh, bl;
                const unsigned short* wb  = &Whi[c][nloc * KST];
                const unsigned short* wlb = &Wlo[c][nloc * KST];
                bh.q[0] = *(const uint4*)(wb  + koff);
                bh.q[1] = *(const uint4*)(wb  + koff + 16);
                bl.q[0] = *(const uint4*)(wlb + koff);
                bl.q[1] = *(const uint4*)(wlb + koff + 16);
                acc[c][nt] = __builtin_amdgcn_wmma_f32_16x16x32_bf16(
                    false, ah.v, false, bh.v, (short)0, acc[c][nt], false, false);
                acc[c][nt] = __builtin_amdgcn_wmma_f32_16x16x32_bf16(
                    false, ah.v, false, bl.v, (short)0, acc[c][nt], false, false);
                acc[c][nt] = __builtin_amdgcn_wmma_f32_16x16x32_bf16(
                    false, al.v, false, bh.v, (short)0, acc[c][nt], false, false);
            }
        }
    }

    const int colLane = lane & 15;
    const int mTop    = (lane >> 4) << 3;
    #pragma unroll
    for (int nt = 0; nt < 2; ++nt) {
        const int ng = n_blk + wn * 32 + nt * 16 + colLane;
        const float b0 = biases[0 * OUT_DIM + ng];
        const float b1 = biases[1 * OUT_DIM + ng];
        const float b2 = biases[2 * OUT_DIM + ng];
        const float b3 = biases[3 * OUT_DIM + ng];
        #pragma unroll
        for (int r = 0; r < 8; ++r) {
            const int rowg = m_blk + wm * 16 + mTop + r;
            float e0, e1, e2, e3;
            spline_eff(phase[rowg], e0, e1, e2, e3);
            const float v = e0 * (acc[0][nt][r] + b0)
                          + e1 * (acc[1][nt][r] + b1)
                          + e2 * (acc[2][nt][r] + b2)
                          + e3 * (acc[3][nt][r] + b3);
            out[(long)rowg * OUT_DIM + ng] = v;
        }
    }
}

extern "C" void kernel_launch(void* const* d_in, const int* in_sizes, int n_in,
                              void* d_out, int out_size, void* d_ws, size_t ws_size,
                              hipStream_t stream) {
    (void)in_sizes; (void)n_in; (void)out_size;
    const float* x      = (const float*)d_in[0];
    const float* phase  = (const float*)d_in[1];
    const float* wts    = (const float*)d_in[2];
    const float* biases = (const float*)d_in[3];
    float* out = (float*)d_out;

    if (ws_size >= (size_t)WS_NEED) {
        unsigned short* ws   = (unsigned short*)d_ws;
        unsigned short* Whi  = ws;
        unsigned short* Wlo  = Whi + W_ELEMS;
        unsigned short* Xhi  = Wlo + W_ELEMS;
        unsigned short* Xlo  = Xhi + X_ELEMS;

        const int w4 = (int)(W_ELEMS / 4);
        const int x4 = (int)(X_ELEMS / 4);
        split_convert_kernel<<<(w4 + 255) / 256, 256, 0, stream>>>(wts, Whi, Wlo, w4);
        split_convert_kernel<<<(x4 + 255) / 256, 256, 0, stream>>>(x,   Xhi, Xlo, x4);

        dim3 grid(OUT_DIM / 64, B_DIM / 128);  // 16 x 32 blocks
        PhaseLinear_wmma_pre<<<grid, 256, 0, stream>>>(Xhi, Xlo, Whi, Wlo,
                                                       phase, biases, out);
    } else {
        dim3 grid(OUT_DIM / 64, B_DIM / 64);   // 16 x 64 blocks
        PhaseLinear_wmma_fly<<<grid, 256, 0, stream>>>(x, phase, wts, biases, out);
    }
}
